// chebVAE_7078106104496
// MI455X (gfx1250) — matrix operand order repaired and checked by reading, not compiled, not run on hardware
//
#include <hip/hip_runtime.h>
#include <hip/hip_bf16.h>

typedef __attribute__((ext_vector_type(16))) _Float16 v16h;
typedef __attribute__((ext_vector_type(8)))  float    v8f;
typedef __attribute__((ext_vector_type(2)))  float    v2f;

#define NGRAPH 512
#define NNODE  64
#define NTOT   32768
#define NEDGE  1048576
#define F_IN   128
#define H1D    256
#define H2D    64
#define KCH    7
#define PADJ   2016

// ---------------- setup kernels ----------------

__global__ __launch_bounds__(256) void zero_f(float* p, int n) {
    int i = blockIdx.x * 256 + threadIdx.x;
    if (i < n) p[i] = 0.f;
}

__global__ __launch_bounds__(256) void build_deg(const int* __restrict__ row,
                                                 float* __restrict__ deg) {
    int e = blockIdx.x * 256 + threadIdx.x;  // grid == NEDGE exactly
    atomicAdd(&deg[row[e]], 1.0f);
}

__global__ __launch_bounds__(256) void calc_dis(const float* __restrict__ deg,
                                                float* __restrict__ dis) {
    int i = blockIdx.x * 256 + threadIdx.x;
    float d = deg[i];
    dis[i] = (d > 0.f) ? rsqrtf(fmaxf(d, 1.f)) : 0.f;
}

__global__ __launch_bounds__(256) void build_lhat(const int* __restrict__ row,
                                                  const int* __restrict__ col,
                                                  const float* __restrict__ dis,
                                                  float* __restrict__ L) {
    int e = blockIdx.x * 256 + threadIdx.x;
    int r = row[e], c = col[e];
    float w = -dis[r] * dis[c];
    atomicAdd(&L[(((size_t)(r >> 6)) << 12) + ((size_t)(r & 63) << 6) + (c & 63)], w);
}

// Repack f32 weights [K][Fdim][Hdim] into WMMA B-operand f16 layout:
// linear index = (((k*KB + kb)*CB + cb)*32 + lane)*16 + e,
// value = W[k][kb*32 + (e>=8?16:0) + ((lane>=16)?8:0) + (e&7)][cb*16 + (lane&15)]
__global__ __launch_bounds__(256) void swizzle_w(const float* __restrict__ W,
                                                 _Float16* __restrict__ out,
                                                 int Fdim, int Hdim, int total) {
    int tid = blockIdx.x * 256 + threadIdx.x;
    if (tid >= total) return;
    int e    = tid & 15;
    int lane = (tid >> 4) & 31;
    int rest = tid >> 9;
    int CB   = Hdim >> 4;
    int cb   = rest % CB;
    int r2   = rest / CB;
    int KB   = Fdim >> 5;
    int kb   = r2 % KB;
    int k    = r2 / KB;
    int Kv   = ((e >> 3) << 4) + ((lane & 16) ? 8 : 0) + (e & 7);
    int f    = kb * 32 + Kv;
    int h    = cb * 16 + (lane & 15);
    out[tid] = (_Float16)W[((size_t)k * Fdim + f) * Hdim + h];
}

// Gather a 16x32 f16 A-tile (row stride `stride`) into WMMA A-register layout.
__device__ __forceinline__ v16h load_a_tile(const _Float16* __restrict__ base,
                                            int stride, int lane) {
    v16h a;
    const _Float16* tr = base + (lane & 15) * stride;
    int hi8 = (lane & 16) ? 8 : 0;
#pragma unroll
    for (int i = 0; i < 8; ++i) {
        int Kb = ((i >> 2) << 4) | hi8 | ((i & 3) << 1);
        a[2 * i]     = tr[Kb];
        a[2 * i + 1] = tr[Kb + 1];
    }
    return a;
}

// Propagation via exact-f32 WMMA: P = sL(64x64) @ Tin(64x64); each wave owns
// 2 output tiles (rows rb*16..+16, cols (cp0,cp0+1)*16). If mode==0 the result
// is stored to Tout; if mode==1 the Chebyshev recurrence Tout = 2*P - Tout is
// applied in place (element-wise, race free).
__device__ __forceinline__ void propagate_wmma(const float* __restrict__ sL,
                                               const float* __restrict__ Tin,
                                               float* __restrict__ Tout,
                                               int rb, int cp0, int lane, int mode) {
    v8f p0 = {0.f, 0.f, 0.f, 0.f, 0.f, 0.f, 0.f, 0.f};
    v8f p1 = p0;
    int Ml   = lane & 15;
    int koff = (lane & 16) ? 2 : 0;
    int arow = ((rb << 4) + Ml) * 64;
    int n0   = (cp0 << 4) + Ml;
    int n1   = n0 + 16;
#pragma unroll
    for (int s = 0; s < 16; ++s) {
        int k4 = (s << 2) + koff;
        v2f a, b0, b1;
        a[0]  = sL[arow + k4];
        a[1]  = sL[arow + k4 + 1];
        b0[0] = Tin[k4 * 64 + n0];
        b0[1] = Tin[(k4 + 1) * 64 + n0];
        b1[0] = Tin[k4 * 64 + n1];
        b1[1] = Tin[(k4 + 1) * 64 + n1];
        p0 = __builtin_amdgcn_wmma_f32_16x16x4_f32(false, a, false, b0, (short)0, p0, false, false);
        p1 = __builtin_amdgcn_wmma_f32_16x16x4_f32(false, a, false, b1, (short)0, p1, false, false);
    }
    int Mb = (rb << 4) + ((lane & 16) ? 8 : 0);
    if (mode == 0) {
#pragma unroll
        for (int r = 0; r < 8; ++r) {
            Tout[(Mb + r) * 64 + n0] = p0[r];
            Tout[(Mb + r) * 64 + n1] = p1[r];
        }
    } else {
#pragma unroll
        for (int r = 0; r < 8; ++r) {
            int i0 = (Mb + r) * 64 + n0;
            int i1 = (Mb + r) * 64 + n1;
            Tout[i0] = 2.f * p0[r] - Tout[i0];
            Tout[i1] = 2.f * p1[r] - Tout[i1];
        }
    }
}

// ---------------- Cheb layer 1: h1 = relu(sum_k T_k(L) x W1[k] + b1) ----------------

__global__ __launch_bounds__(256) void cheb1_kernel(const float* __restrict__ x,
                                                    const float* __restrict__ Lhat,
                                                    const _Float16* __restrict__ W1h,
                                                    const float* __restrict__ b1,
                                                    float* __restrict__ h1) {
    __shared__ float    sL[64 * 64];      // 16 KB  L_hat for this graph
    __shared__ float    sTa[64 * 64];     // 16 KB  Chebyshev buffer
    __shared__ float    sTb[64 * 64];     // 16 KB  Chebyshev buffer
    __shared__ _Float16 sTh[64 * 64];     //  8 KB  f16 staging of current T_k chunk

    int g    = blockIdx.x;
    int tid  = threadIdx.x;
    int lane = tid & 31;
    int wave = tid >> 5;
    int rb   = wave >> 1;            // row block (0..3), shared by both phases
    int cb0  = (wave & 1) * 8;       // f16-phase col block base (8 tiles per wave)
    int cp0  = (wave & 1) * 2;       // propagation-phase col block base (2 tiles)

    // load L_hat
    {
        const float4* Lg  = (const float4*)(Lhat + (size_t)g * 4096);
        float4*       sL4 = (float4*)sL;
        for (int i = tid; i < 1024; i += 256) sL4[i] = Lg[i];
    }

    v8f zero8 = {0.f, 0.f, 0.f, 0.f, 0.f, 0.f, 0.f, 0.f};
    v8f acc[8];
#pragma unroll
    for (int t = 0; t < 8; ++t) acc[t] = zero8;

    for (int c = 0; c < 2; ++c) {   // feature chunks of 64 (F_IN = 128)
        // load x chunk -> sTa  (64 rows x 64 cols)
        const float* xg = x + (size_t)g * 64 * F_IN + c * 64;
        for (int i = tid; i < 1024; i += 256) {
            int m = i >> 4, f4 = (i & 15) << 2;
            *(float4*)&sTa[m * 64 + f4] = *(const float4*)&xg[m * F_IN + f4];
        }
        __syncthreads();

        float* Tpp = sTa;   // T_{k-2}
        float* Tp  = sTb;   // T_{k-1}

        for (int k = 0; k < KCH; ++k) {
            float* cur;
            if (k == 0) {
                cur = sTa;
            } else if (k == 1) {
                propagate_wmma(sL, sTa, sTb, rb, cp0, lane, 0);
                cur = sTb;
                __syncthreads();
            } else {
                propagate_wmma(sL, Tp, Tpp, rb, cp0, lane, 1);
                cur = Tpp;
                float* tsw = Tpp; Tpp = Tp; Tp = tsw;
                __syncthreads();
            }
            // convert current T_k chunk to f16
            for (int i = tid; i < 4096; i += 256) sTh[i] = (_Float16)cur[i];
            __syncthreads();

            // WMMA accumulate: acc += A(T_k tile) * B(W1[k] tile)
            for (int kb = 0; kb < 2; ++kb) {
                v16h a = load_a_tile(&sTh[(rb << 4) * 64 + kb * 32], 64, lane);
                int kb2 = c * 2 + kb;
                const v16h* Wv = (const v16h*)W1h;
#pragma unroll
                for (int t = 0; t < 8; ++t) {
                    int cb = cb0 + t;
                    v16h b = Wv[(size_t)(((k * 4 + kb2) * 16 + cb) * 32 + lane)];
                    acc[t] = __builtin_amdgcn_wmma_f32_16x16x32_f16(
                        false, a, false, b, (short)0, acc[t], false, false);
                }
            }
            __syncthreads();
        }
    }

    // epilogue: bias + relu + store h1
    {
        int Nl = lane & 15;
        int Mb = (rb << 4) + ((lane & 16) ? 8 : 0);
#pragma unroll
        for (int t = 0; t < 8; ++t) {
            int Nc = (cb0 + t) * 16 + Nl;
            float bv = b1[Nc];
#pragma unroll
            for (int r = 0; r < 8; ++r) {
                float v = acc[t][r] + bv;
                h1[((size_t)(g * 64) + Mb + r) * H1D + Nc] = fmaxf(v, 0.f);
            }
        }
    }
}

// -------- Cheb layers 2+3 (shared propagation): mu, logvar per graph --------

__global__ __launch_bounds__(256) void cheb23_kernel(const float* __restrict__ h1,
                                                     const float* __restrict__ Lhat,
                                                     const _Float16* __restrict__ W2h,
                                                     const _Float16* __restrict__ W3h,
                                                     const float* __restrict__ b2,
                                                     const float* __restrict__ b3,
                                                     float* __restrict__ mu_out,
                                                     float* __restrict__ lv_out) {
    __shared__ float    sL[64 * 64];
    __shared__ float    sSa[64 * 64];
    __shared__ float    sSb[64 * 64];
    __shared__ _Float16 sSh[64 * 64];

    int g    = blockIdx.x;
    int tid  = threadIdx.x;
    int lane = tid & 31;
    int wave = tid >> 5;
    int rb   = wave >> 1;
    int cbs  = (wave & 1) * 2;   // 2 mu + 2 lv tiles per wave (H2 = 64 -> 4 col blocks)
    int cp0  = (wave & 1) * 2;   // propagation col blocks

    {
        const float4* Lg  = (const float4*)(Lhat + (size_t)g * 4096);
        float4*       sL4 = (float4*)sL;
        for (int i = tid; i < 1024; i += 256) sL4[i] = Lg[i];
    }

    v8f zero8 = {0.f, 0.f, 0.f, 0.f, 0.f, 0.f, 0.f, 0.f};
    v8f accm[2], accl[2];
#pragma unroll
    for (int t = 0; t < 2; ++t) { accm[t] = zero8; accl[t] = zero8; }

    for (int c = 0; c < 4; ++c) {   // H1 = 256 -> 4 chunks of 64
        const float* hg = h1 + (size_t)g * 64 * H1D + c * 64;
        for (int i = tid; i < 1024; i += 256) {
            int m = i >> 4, f4 = (i & 15) << 2;
            *(float4*)&sSa[m * 64 + f4] = *(const float4*)&hg[m * H1D + f4];
        }
        __syncthreads();

        float* Tpp = sSa;
        float* Tp  = sSb;

        for (int k = 0; k < KCH; ++k) {
            float* cur;
            if (k == 0) {
                cur = sSa;
            } else if (k == 1) {
                propagate_wmma(sL, sSa, sSb, rb, cp0, lane, 0);
                cur = sSb;
                __syncthreads();
            } else {
                propagate_wmma(sL, Tp, Tpp, rb, cp0, lane, 1);
                cur = Tpp;
                float* tsw = Tpp; Tpp = Tp; Tp = tsw;
                __syncthreads();
            }
            for (int i = tid; i < 4096; i += 256) sSh[i] = (_Float16)cur[i];
            __syncthreads();

            for (int kb = 0; kb < 2; ++kb) {
                v16h a = load_a_tile(&sSh[(rb << 4) * 64 + kb * 32], 64, lane);
                int kb2 = c * 2 + kb;
                const v16h* W2v = (const v16h*)W2h;
                const v16h* W3v = (const v16h*)W3h;
#pragma unroll
                for (int t = 0; t < 2; ++t) {
                    int cb = cbs + t;
                    size_t bi = (size_t)(((k * 8 + kb2) * 4 + cb) * 32 + lane);
                    v16h bm = W2v[bi];
                    accm[t] = __builtin_amdgcn_wmma_f32_16x16x32_f16(
                        false, a, false, bm, (short)0, accm[t], false, false);
                    v16h bl = W3v[bi];
                    accl[t] = __builtin_amdgcn_wmma_f32_16x16x32_f16(
                        false, a, false, bl, (short)0, accl[t], false, false);
                }
            }
            __syncthreads();
        }
    }

    // write node-level tiles to LDS, then graph-mean reduce
    {
        int Nl = lane & 15;
        int Mb = (rb << 4) + ((lane & 16) ? 8 : 0);
#pragma unroll
        for (int t = 0; t < 2; ++t) {
            int cb = cbs + t;
#pragma unroll
            for (int r = 0; r < 8; ++r) {
                sSa[(Mb + r) * 64 + cb * 16 + Nl] = accm[t][r];
                sSb[(Mb + r) * 64 + cb * 16 + Nl] = accl[t][r];
            }
        }
    }
    __syncthreads();
    if (tid < 64) {
        float s = 0.f;
        for (int m = 0; m < 64; ++m) s += sSa[m * 64 + tid];
        mu_out[(size_t)g * 64 + tid] = s * 0.015625f + b2[tid];
    } else if (tid < 128) {
        int hcol = tid - 64;
        float s = 0.f;
        for (int m = 0; m < 64; ++m) s += sSb[m * 64 + hcol];
        lv_out[(size_t)g * 64 + hcol] = s * 0.015625f + b3[hcol];
    }
}

// --------- decoder: z = mu + eps*exp(lv/2); adj = sig(relu(z@Wf1+bf1)@Wf2+bf2) ---------
// One workgroup per 16-graph row block; both decoder GEMMs on f16 WMMA.

__global__ __launch_bounds__(256) void decoder_kernel(const float* __restrict__ mu,
                                                      const float* __restrict__ lv,
                                                      const float* __restrict__ eps,
                                                      const _Float16* __restrict__ Wf1h,
                                                      const float* __restrict__ bf1,
                                                      const _Float16* __restrict__ Wf2h,
                                                      const float* __restrict__ bf2,
                                                      float* __restrict__ adj) {
    __shared__ _Float16 sZh[16 * 64];    // 2 KB  z row block (f16)
    __shared__ _Float16 sHh[16 * 256];   // 8 KB  hidden row block (f16)

    int grb  = blockIdx.x;               // graph row block (16 graphs)
    int tid  = threadIdx.x;
    int lane = tid & 31;
    int wave = tid >> 5;
    int Nl   = lane & 15;
    int hi8  = (lane & 16) ? 8 : 0;
    v8f zero8 = {0.f, 0.f, 0.f, 0.f, 0.f, 0.f, 0.f, 0.f};

    // z = mu + eps * exp(0.5*logvar), staged as f16
    for (int j = tid; j < 16 * 64; j += 256) {
        int row = j >> 6, col = j & 63;
        size_t gi = ((size_t)(grb * 16 + row)) * 64 + col;
        float m = mu[gi], l = lv[gi];
        sZh[j] = (_Float16)(m + eps[gi] * expf(0.5f * l));
    }
    __syncthreads();

    // H = relu(z @ Wf1 + bf1): 16x256, 2 col tiles per wave, K = 64 (2 blocks)
    {
        v16h aZ[2];
#pragma unroll
        for (int kb = 0; kb < 2; ++kb) aZ[kb] = load_a_tile(&sZh[kb * 32], 64, lane);
        const v16h* Wv = (const v16h*)Wf1h;
#pragma unroll
        for (int t = 0; t < 2; ++t) {
            int cb = wave * 2 + t;
            v8f acc = zero8;
#pragma unroll
            for (int kb = 0; kb < 2; ++kb) {
                v16h b = Wv[(size_t)((kb * 16 + cb) * 32 + lane)];
                acc = __builtin_amdgcn_wmma_f32_16x16x32_f16(
                    false, aZ[kb], false, b, (short)0, acc, false, false);
            }
            int Nc = cb * 16 + Nl;
            float bv = bf1[Nc];
#pragma unroll
            for (int r = 0; r < 8; ++r)
                sHh[(hi8 + r) * 256 + Nc] = (_Float16)fmaxf(acc[r] + bv, 0.f);
        }
    }
    __syncthreads();

    // adj = sigmoid(H @ Wf2 + bf2): 16x2016, K = 256 (8 blocks), 126 col tiles
    {
        v16h aH[8];
#pragma unroll
        for (int kb = 0; kb < 8; ++kb) aH[kb] = load_a_tile(&sHh[kb * 32], 256, lane);
        const v16h* Wv = (const v16h*)Wf2h;
        for (int cb = wave; cb < 126; cb += 8) {
            v8f acc = zero8;
#pragma unroll
            for (int kb = 0; kb < 8; ++kb) {
                v16h b = Wv[(size_t)((kb * 126 + cb) * 32 + lane)];
                acc = __builtin_amdgcn_wmma_f32_16x16x32_f16(
                    false, aH[kb], false, b, (short)0, acc, false, false);
            }
            int Nc = cb * 16 + Nl;
            float bv = bf2[Nc];
#pragma unroll
            for (int r = 0; r < 8; ++r) {
                float v = acc[r] + bv;
                adj[((size_t)(grb * 16) + hi8 + r) * PADJ + Nc] = 1.f / (1.f + expf(-v));
            }
        }
    }
}

// ---------------- launch ----------------

extern "C" void kernel_launch(void* const* d_in, const int* in_sizes, int n_in,
                              void* d_out, int out_size, void* d_ws, size_t ws_size,
                              hipStream_t stream) {
    (void)in_sizes; (void)n_in; (void)out_size; (void)ws_size;
    const float* x   = (const float*)d_in[0];
    const int*   ei  = (const int*)d_in[1];
    const float* eps = (const float*)d_in[3];
    const float* W1  = (const float*)d_in[4];
    const float* b1  = (const float*)d_in[5];
    const float* W2  = (const float*)d_in[6];
    const float* b2  = (const float*)d_in[7];
    const float* W3  = (const float*)d_in[8];
    const float* b3  = (const float*)d_in[9];
    const float* Wf1 = (const float*)d_in[10];
    const float* bf1 = (const float*)d_in[11];
    const float* Wf2 = (const float*)d_in[12];
    const float* bf2 = (const float*)d_in[13];

    char* ws = (char*)d_ws;
    float*    deg  = (float*)(ws + 0);            // 131072 B
    float*    dis  = (float*)(ws + 131072);       // 131072 B
    float*    Lh   = (float*)(ws + 262144);       // 8 MB
    float*    h1   = (float*)(ws + 8650752);      // 32 MB
    _Float16* W1h  = (_Float16*)(ws + 42205184);  // 448 KB
    _Float16* W2h  = (_Float16*)(ws + 42663936);  // 224 KB
    _Float16* W3h  = (_Float16*)(ws + 42893312);  // 224 KB
    _Float16* Wf1h = (_Float16*)(ws + 43122688);  // 32 KB
    _Float16* Wf2h = (_Float16*)(ws + 43155456);  // 1008 KB  (end ~44.2 MB)

    float* adj    = (float*)d_out;
    float* mu_out = adj + (size_t)NGRAPH * PADJ;
    float* lv_out = mu_out + (size_t)NGRAPH * H2D;

    // zero deg, dis, Lhat region (2,162,688 floats)
    zero_f<<<8448, 256, 0, stream>>>((float*)ws, 2162688);
    build_deg<<<NEDGE / 256, 256, 0, stream>>>(ei, deg);
    calc_dis<<<NTOT / 256, 256, 0, stream>>>(deg, dis);
    build_lhat<<<NEDGE / 256, 256, 0, stream>>>(ei, ei + NEDGE, dis, Lh);

    swizzle_w<<<896, 256, 0, stream>>>(W1, W1h, F_IN, H1D, 229376);
    swizzle_w<<<448, 256, 0, stream>>>(W2, W2h, H1D, H2D, 114688);
    swizzle_w<<<448, 256, 0, stream>>>(W3, W3h, H1D, H2D, 114688);
    swizzle_w<<<64, 256, 0, stream>>>(Wf1, Wf1h, H2D, H1D, 16384);
    swizzle_w<<<2016, 256, 0, stream>>>(Wf2, Wf2h, H1D, PADJ, 516096);

    cheb1_kernel<<<NGRAPH, 256, 0, stream>>>(x, Lh, W1h, b1, h1);
    cheb23_kernel<<<NGRAPH, 256, 0, stream>>>(h1, Lh, W2h, W3h, b2, b3, mu_out, lv_out);
    decoder_kernel<<<NGRAPH / 16, 256, 0, stream>>>(mu_out, lv_out, eps,
                                                    Wf1h, bf1, Wf2h, bf2, adj);
}
